// RegularERM_40922448396412
// MI455X (gfx1250) — matrix operand order repaired and checked by reading, not compiled, not run on hardware
//
#include <hip/hip_runtime.h>
#include <hip/hip_bf16.h>
#include <math.h>

#define E_ 4
#define B_ 8192
#define F_ 2048
#define L_ 100
#define NCB 7              // 7 blocks of 16 labels -> 112 padded
#define KC_ 64             // F_/32 K-chunks
#define LAM_ 1.0f

typedef __attribute__((ext_vector_type(16))) _Float16 v16h;
typedef __attribute__((ext_vector_type(8)))  float    v8f;
typedef __attribute__((ext_vector_type(4)))  int      v4i;

// 16-bit A-operand (16x32) K index for half j in lane group hi (ISA 7.12.2)
static __device__ __forceinline__ int kmapA(int j, int hi) {
  return hi * 8 + (j < 8 ? j : j + 8);
}

__global__ __launch_bounds__(256) void k_zero(float* p, int n) {
  int t = blockIdx.x * 256 + threadIdx.x;
  if (t < n) p[t] = 0.0f;
}

// Prepack W (f32 [F,L] row-major) into f16 B-operand lane order:
// w16[((c*64+kc)*32+lane)*16 + j] = f16(W[kc*32 + (lane>>4)*16 + j][16c + (lane&15)])
__global__ __launch_bounds__(256) void k_pack(const float* __restrict__ W,
                                              _Float16* __restrict__ w16) {
  int t = blockIdx.x * 256 + threadIdx.x;
  if (t >= NCB * KC_ * 32 * 16) return;
  int j = t & 15, lane = (t >> 4) & 31, kc = (t >> 9) & 63, c = t >> 15;
  int k = kc * 32 + (lane >> 4) * 16 + j;
  int l = c * 16 + (lane & 15);
  w16[t] = (l < L_) ? (_Float16)W[k * L_ + l] : (_Float16)0.0f;
}

__global__ __launch_bounds__(256) void k_counts(const int* __restrict__ y,
                                                float* __restrict__ counts) {
  int t = blockIdx.x * 256 + threadIdx.x;
  if (t < E_ * B_) atomicAdd(&counts[(t / B_) * L_ + y[t]], 1.0f);
}

// sums[e,l,f] = sum_b onehot[e,b,l] * feats[e,b,f]  via WMMA.
// Onehot A-tiles are built ONCE per block in LDS by sparse scatter (set/clear),
// shared by all 8 waves; A-operand comes from ds_read_b128 pairs.
// grid: 4 envs x 16 f-tiles(128) x 8 b-chunks(1024) = 512 blocks
__global__ __launch_bounds__(256) void k_labelsums(const float* __restrict__ feats,
                                                   const int* __restrict__ y,
                                                   float* __restrict__ sums) {
  __shared__ __align__(32) _Float16 alds[NCB * 32 * 16];  // 7 KB onehot tiles
  int bx = blockIdx.x;
  int e  = bx >> 7;
  int ft = (bx >> 3) & 15;
  int bb = bx & 7;
  int wave = threadIdx.x >> 5, lane = threadIdx.x & 31;
  int hi = lane >> 4, n = lane & 15;
  int fcol = ft * 128 + wave * 16 + n;
  const float* fb = feats + (size_t)e * B_ * F_ + fcol;
  const int*   yb = y + e * B_ + bb * 1024;

  // zero the onehot tiles once
  {
    int* az = (int*)alds;
    for (int i = threadIdx.x; i < NCB * 32 * 8; i += 256) az[i] = 0;
  }
  __syncthreads();

  v8f zero = {};
  v8f acc[NCB];
#pragma unroll
  for (int r = 0; r < NCB; ++r) acc[r] = zero;

  for (int b0 = 0; b0 < 1024; b0 += 32) {
    // scatter-set: one thread per k writes 1.0h into tile (y>>4) at (m=y&15, k)
    int off = 0;
    if (threadIdx.x < 32) {
      int k = threadIdx.x;
      int label = yb[b0 + k];
      int khi = (k >> 3) & 1;                                // A-layout inverse
      int kj = (k < 8) ? k : ((k < 24) ? k - 8 : k - 16);
      off = ((label >> 4) * 32 + (label & 15) + 16 * khi) * 16 + kj;
      alds[off] = (_Float16)1.0f;
    }
    __syncthreads();
    // B operand: feats[k = b0+kmapA(j,hi)][fcol], NT (streamed once), cvt f32->f16
    v16h bm;
#pragma unroll
    for (int j = 0; j < 16; ++j)
      bm[j] = (_Float16)__builtin_nontemporal_load(
          &fb[(size_t)(bb * 1024 + b0 + kmapA(j, hi)) * F_]);
    // A operand: read the 7 onehot tiles straight from LDS
    v16h am[NCB];
#pragma unroll
    for (int r = 0; r < NCB; ++r)
      am[r] = *(const v16h*)(alds + (r * 32 + lane) * 16);
#pragma unroll
    for (int r = 0; r < NCB; ++r)
      acc[r] = __builtin_amdgcn_wmma_f32_16x16x32_f16(false, am[r], false, bm,
                                                      (short)0, acc[r], false, false);
    __syncthreads();
    // scatter-clear (same offsets) -> LDS is all-zero again; program order of
    // this wave guarantees clear precedes next chunk's set, and other waves
    // block at the next post-set barrier.
    if (threadIdx.x < 32) alds[off] = (_Float16)0.0f;
  }
  // D layout: n = lane&15, m = t + hi*8
#pragma unroll
  for (int r = 0; r < NCB; ++r)
#pragma unroll
    for (int t = 0; t < 8; ++t) {
      int l = r * 16 + hi * 8 + t;
      if (l < L_) atomicAdd(&sums[((size_t)e * L_ + l) * F_ + fcol], acc[r][t]);
    }
}

// logits = feats @ W + b, then per-row log-softmax + gather, accumulate CE sum
// grid: 256 blocks x 8 waves, each wave owns 16 rows
__global__ __launch_bounds__(256) void k_ce(const float* __restrict__ feats,
                                            const int* __restrict__ y,
                                            const _Float16* __restrict__ w16,
                                            const float* __restrict__ bias,
                                            float* __restrict__ ce_acc) {
  __shared__ __align__(32) _Float16 wlds[NCB * 32 * 16];  // 7 KB
  int rowbase = blockIdx.x * 128;
  int wave = threadIdx.x >> 5, lane = threadIdx.x & 31;
  int hi = lane >> 4, n = lane & 15;
  const float* arow = feats + (size_t)(rowbase + wave * 16 + n) * F_;

  v8f zero = {};
  v8f acc[NCB];
#pragma unroll
  for (int c = 0; c < NCB; ++c) acc[c] = zero;

  for (int kc = 0; kc < KC_; ++kc) {
    // stage this K-chunk of prepacked W into LDS (7 * 1KB)
    if (threadIdx.x < NCB * 32) {
      int c = threadIdx.x >> 5, ll = threadIdx.x & 31;
      const v4i* src = (const v4i*)(w16 + ((size_t)(c * KC_ + kc) * 32 + ll) * 16);
      v4i* dst = (v4i*)(wlds + (c * 32 + ll) * 16);
      dst[0] = src[0];
      dst[1] = src[1];
    }
    __syncthreads();
    int k0 = kc * 32;
    // A operand: 16 contiguous-run f32 NT loads, cvt to f16
    v16h am;
#pragma unroll
    for (int j = 0; j < 16; ++j)
      am[j] = (_Float16)__builtin_nontemporal_load(&arow[k0 + hi * 16 + j]);
    // B tiles from LDS, then WMMA burst
    v16h bm[NCB];
#pragma unroll
    for (int c = 0; c < NCB; ++c)
      bm[c] = *(const v16h*)(wlds + (c * 32 + lane) * 16);
#pragma unroll
    for (int c = 0; c < NCB; ++c)
      acc[c] = __builtin_amdgcn_wmma_f32_16x16x32_f16(false, am, false, bm[c],
                                                      (short)0, acc[c], false, false);
    __syncthreads();
  }

  float bl[NCB];
#pragma unroll
  for (int c = 0; c < NCB; ++c) {
    int l = c * 16 + n;
    bl[c] = (l < L_) ? bias[l] : 0.0f;
  }

#pragma unroll
  for (int t = 0; t < 8; ++t) {
    int r = rowbase + wave * 16 + hi * 8 + t;  // this 16-lane group's row
    int yr = y[r];
    float v[NCB];
#pragma unroll
    for (int c = 0; c < NCB; ++c) {
      int l = c * 16 + n;
      float x = acc[c][t] + bl[c];
      v[c] = (l < L_) ? x : -__builtin_inff();
    }
    float mx = v[0];
#pragma unroll
    for (int c = 1; c < NCB; ++c) mx = fmaxf(mx, v[c]);
#pragma unroll
    for (int d = 1; d < 16; d <<= 1) mx = fmaxf(mx, __shfl_xor(mx, d, 16));
    float s = 0.0f;
#pragma unroll
    for (int c = 0; c < NCB; ++c) s += __expf(v[c] - mx);
#pragma unroll
    for (int d = 1; d < 16; d <<= 1) s += __shfl_xor(s, d, 16);
    float ly = 0.0f;
    if (n == (yr & 15)) {
      int cy = yr >> 4;
      float pick = v[0];
#pragma unroll
      for (int c = 1; c < NCB; ++c)
        if (cy == c) pick = v[c];
      ly = pick;
    }
#pragma unroll
    for (int d = 1; d < 16; d <<= 1) ly += __shfl_xor(ly, d, 16);
    if (n == 0) atomicAdd(ce_acc, mx + __logf(s) - ly);
  }
}

__global__ __launch_bounds__(256) void k_final(const float* __restrict__ sums,
                                               const float* __restrict__ counts,
                                               const float* __restrict__ ce_acc,
                                               float* __restrict__ out) {
  __shared__ float varsh[E_ * L_];
  __shared__ float red[16];
  int wave = threadIdx.x >> 5, lane = threadIdx.x & 31;
  for (int p = wave; p < E_ * L_; p += 8) {
    const float* sp = sums + (size_t)p * F_;
    float inv = 1.0f / fmaxf(counts[p], 1.0f);
    float s1 = 0.f, s2 = 0.f;
    for (int f = lane; f < F_; f += 32) {
      float m = sp[f] * inv;
      s1 += m;
      s2 += m * m;
    }
#pragma unroll
    for (int d = 1; d < 32; d <<= 1) {
      s1 += __shfl_xor(s1, d, 32);
      s2 += __shfl_xor(s2, d, 32);
    }
    if (lane == 0) {
      float mu = s1 / (float)F_;
      varsh[p] = (s2 - (float)F_ * mu * mu) / (float)(F_ - 1);
    }
  }
  __syncthreads();
  float contrib = 0.f, selc = 0.f;
  if (threadIdx.x < L_) {
    int l = threadIdx.x;
    float nvis = 0.f, vsum = 0.f;
#pragma unroll
    for (int e = 0; e < E_; ++e) {
      float c = counts[e * L_ + l];
      float vis = (c > 0.f) ? 1.f : 0.f;
      nvis += vis;
      vsum += vis * varsh[e * L_ + l];
    }
    float per = vsum / fmaxf(nvis, 1.f);
    float sel = (nvis > 1.5f) ? 1.f : 0.f;
    contrib = per * sel;
    selc = sel;
  }
#pragma unroll
  for (int d = 1; d < 32; d <<= 1) {
    contrib += __shfl_xor(contrib, d, 32);
    selc += __shfl_xor(selc, d, 32);
  }
  if (lane == 0) {
    red[wave] = contrib;
    red[8 + wave] = selc;
  }
  __syncthreads();
  if (threadIdx.x == 0) {
    float tot = 0.f, sc = 0.f;
#pragma unroll
    for (int w = 0; w < 8; ++w) {
      tot += red[w];
      sc += red[8 + w];
    }
    float mean_loss = LAM_ * tot / fmaxf(sc, 1.f);
    float ce = ce_acc[0] / (float)(E_ * B_);
    out[0] = mean_loss + ce;
  }
}

extern "C" void kernel_launch(void* const* d_in, const int* in_sizes, int n_in,
                              void* d_out, int out_size, void* d_ws, size_t ws_size,
                              hipStream_t stream) {
  (void)in_sizes; (void)n_in; (void)out_size; (void)ws_size;
  const float* feats = (const float*)d_in[0];
  const int*   y     = (const int*)d_in[1];
  const float* W     = (const float*)d_in[2];
  const float* bias  = (const float*)d_in[3];
  float* out = (float*)d_out;

  // workspace layout (16B-aligned sections):
  float* sums   = (float*)d_ws;                       // E*L*F
  float* counts = sums + (size_t)E_ * L_ * F_;        // E*L
  float* ce     = counts + E_ * L_;                   // 1 (+3 pad)
  _Float16* w16 = (_Float16*)(ce + 4);                // NCB*64*32*16 halfs

  int nz = E_ * L_ * F_ + E_ * L_ + 4;
  k_zero<<<(nz + 255) / 256, 256, 0, stream>>>((float*)d_ws, nz);
  k_pack<<<(NCB * KC_ * 32 * 16 + 255) / 256, 256, 0, stream>>>(W, w16);
  k_counts<<<(E_ * B_) / 256, 256, 0, stream>>>(y, counts);
  k_labelsums<<<E_ * 16 * 8, 256, 0, stream>>>(feats, y, sums);
  k_ce<<<(E_ * B_) / 128, 256, 0, stream>>>(feats, y, w16, bias, ce);
  k_final<<<1, 256, 0, stream>>>(sums, counts, ce, out);
}
